// V1SparseCoding_74801150427780
// MI455X (gfx1250) — matrix operand order repaired and checked by reading, not compiled, not run on hardware
//
#include <hip/hip_runtime.h>
#include <hip/hip_bf16.h>
#include <math.h>

typedef __attribute__((ext_vector_type(16))) __bf16 v16bf;
typedef __attribute__((ext_vector_type(8)))  float  v8f;

#define N_BASES       256
#define INPUT_DIM     768
#define PPI           729     // patches per image (27*27)
#define N_ROWS        46656   // 64 * 729
#define ROW_TILES     2916    // N_ROWS / 16
#define KCH_B         24      // 768/32 K-chunks for feedforward GEMM
#define KCH_G         8       // 256/32 K-chunks for LCA GEMM
#define NTILES        16      // 256/16 column tiles
#define LAMBDA        0.5f
#define DT_OVER_TAU   0.01f
#define LCA_ITERS     50

union Frag { unsigned int u[8]; v16bf v; };
union AccU { v8f v; float4 f4[2]; float f[8]; };

// float -> bf16 round-to-nearest-even
__device__ __forceinline__ unsigned short f2bf(float x) {
    unsigned int b = __builtin_bit_cast(unsigned int, x);
    unsigned int r = b + 0x7FFFu + ((b >> 16) & 1u);
    return (unsigned short)(r >> 16);
}

__device__ __forceinline__ float softshrink(float u) {
    return copysignf(fmaxf(fabsf(u) - LAMBDA, 0.0f), u);
}

__device__ __forceinline__ v8f wmma_bf16(const Frag& A, const Frag& B, v8f C) {
    return __builtin_amdgcn_wmma_f32_16x16x32_bf16(false, A.v, false, B.v,
                                                   (short)0, C, false, false);
}

// A-matrix (16x32 bf16) fragment local ushort index for element (row m, k-in-chunk kin)
// Layout (ISA 7.12.2): VGPR0-3 lanes0-15:K0-7 lanes16-31:K8-15; VGPR4-7: +16
__device__ __forceinline__ int a_local(int m, int kin) {
    int group  = kin >> 4;        // 0: K0-15, 1: K16-31
    int within = kin & 15;
    int half   = within >> 3;     // lane half
    int off    = within & 7;
    int v      = (group << 2) + (off >> 1);
    int pair   = off & 1;
    int lane   = (half << 4) + m;
    return (lane * 8 + v) * 2 + pair;   // ushort index within 512-ushort block
}

// B-matrix (32x16 bf16) fragment local ushort index for (k-in-chunk kin, col jin)
// Layout: VGPR0-3 lanes0-15:K0-7 lanes16-31:K8-15; VGPR4-7: +16
__device__ __forceinline__ int b_local(int kin, int jin) {
    int khalf = (kin >> 3) & 1;
    int v     = ((kin >> 4) << 2) + ((kin & 7) >> 1);
    int pair  = kin & 1;
    int lane  = (khalf << 4) + jin;
    return (lane * 8 + v) * 2 + pair;
}

// ---------------- gram = D D^T - I, packed to bf16 B-fragment layout ----------------
__global__ void gram_pack_kernel(const float* __restrict__ D,
                                 unsigned short* __restrict__ gramB16) {
    __shared__ float drow[INPUT_DIM];
    int i = blockIdx.x, j = threadIdx.x;
    for (int x = j; x < INPUT_DIM; x += 256) drow[x] = D[(size_t)i * INPUT_DIM + x];
    __syncthreads();
    const float* dj = D + (size_t)j * INPUT_DIM;
    float g = 0.0f;
    #pragma unroll 4
    for (int k = 0; k < INPUT_DIM; k++) g = fmaf(drow[k], dj[k], g);
    if (i == j) g -= 1.0f;
    int kchunk = i >> 5, kin = i & 31, ntile = j >> 4, jin = j & 15;
    gramB16[((size_t)kchunk * NTILES + ntile) * 512 + b_local(kin, jin)] = f2bf(g);
}

// ---------------- D^T packed to bf16 B-fragment layout (for b = pf @ D^T) ----------------
__global__ void dict_pack_kernel(const float* __restrict__ D,
                                 unsigned short* __restrict__ dictB16) {
    int k = blockIdx.x, j = threadIdx.x;           // B[k][j] = D[j][k]
    float val = D[(size_t)j * INPUT_DIM + k];
    int kchunk = k >> 5, kin = k & 31, ntile = j >> 4, jin = j & 15;
    dictB16[((size_t)kchunk * NTILES + ntile) * 512 + b_local(kin, jin)] = f2bf(val);
}

// ---------------- extract + standardize patches, scatter into A-fragment layout ----------------
__global__ void patch_std_kernel(const float* __restrict__ img,
                                 unsigned short* __restrict__ pfA16) {
    __shared__ float rs[256], rq[256];
    int p = blockIdx.x, t = threadIdx.x;
    int imgIdx = p / PPI;
    int rem    = p % PPI;
    int ph = rem / 27, pw = rem % 27;
    int py = t >> 4, px = t & 15;
    int row = ph * 8 + py, col = pw * 8 + px;
    float v[3]; float s = 0.0f, q = 0.0f;
    #pragma unroll
    for (int c = 0; c < 3; c++) {
        float x = img[(((size_t)imgIdx * 3 + c) * 224 + row) * 224 + col];
        v[c] = x; s += x; q += x * x;
    }
    rs[t] = s; rq[t] = q; __syncthreads();
    for (int o = 128; o > 0; o >>= 1) {
        if (t < o) { rs[t] += rs[t + o]; rq[t] += rq[t + o]; }
        __syncthreads();
    }
    float mean = rs[0] * (1.0f / 768.0f);
    float var  = (rq[0] - 768.0f * mean * mean) * (1.0f / 767.0f);  // ddof=1
    float inv  = 1.0f / (sqrtf(fmaxf(var, 0.0f)) + 1e-8f);
    int tile = p >> 4, m = p & 15;
    #pragma unroll
    for (int c = 0; c < 3; c++) {
        int f = c * 256 + t;          // feature index in (C, ph, pw) order
        float x = (v[c] - mean) * inv;
        int kchunk = f >> 5, kin = f & 31;
        pfA16[((size_t)tile * KCH_B + kchunk) * 512 + a_local(m, kin)] = f2bf(x);
    }
}

// ---------------- out init to -inf for max pooling ----------------
__global__ void init_out_kernel(float* __restrict__ out) {
    int i = blockIdx.x * 256 + threadIdx.x;
    if (i < 64 * N_BASES) out[i] = -3.402823466e38f;
}

// ---------------- fused feedforward GEMM + 50-iteration LCA ----------------
// block = 256 threads (8 waves); block covers 16 patch rows x 256 codes.
// u, b, acc live in WMMA C-fragments; gram B-fragments are register-resident
// (loop-invariant, 128 VGPRs); only an 8 KB LDS buffer stages the shared
// A-matrix (softshrink(u) in bf16) each iteration.
__global__ void lca_kernel(const unsigned int* __restrict__ pfA,
                           const unsigned int* __restrict__ dictB,
                           const unsigned int* __restrict__ gramB,
                           float* __restrict__ out) {
    __shared__ __align__(16) unsigned int aPack[2048];   // 8 KB A staging
    unsigned short* aPack16 = (unsigned short*)aPack;

    int t = threadIdx.x, lane = t & 31, wave = t >> 5;
    int tile = blockIdx.x;
    int nt0 = wave * 2, nt1 = wave * 2 + 1;

    // --- prologue: b = pf @ D^T over K = 768 (48 WMMAs per wave) ---
    v8f b0 = {}, b1 = {};
    #pragma unroll 4
    for (int kc = 0; kc < KCH_B; kc++) {
        Frag A, B0, B1;
        const uint4* ap = (const uint4*)(pfA + (((size_t)tile * KCH_B + kc) * 32 + lane) * 8);
        *(uint4*)&A.u[0] = ap[0]; *(uint4*)&A.u[4] = ap[1];
        const uint4* b0p = (const uint4*)(dictB + (((size_t)kc * NTILES + nt0) * 32 + lane) * 8);
        *(uint4*)&B0.u[0] = b0p[0]; *(uint4*)&B0.u[4] = b0p[1];
        const uint4* b1p = (const uint4*)(dictB + (((size_t)kc * NTILES + nt1) * 32 + lane) * 8);
        *(uint4*)&B1.u[0] = b1p[0]; *(uint4*)&B1.u[4] = b1p[1];
        b0 = wmma_bf16(A, B0, b0);
        b1 = wmma_bf16(A, B1, b1);
    }

    // --- preload gram B-fragments: loop-invariant, register-resident (128 VGPRs) ---
    Frag G0[KCH_G], G1[KCH_G];
    #pragma unroll
    for (int kc = 0; kc < KCH_G; kc++) {
        const uint4* g0p = (const uint4*)(gramB + (((size_t)kc * NTILES + nt0) * 32 + lane) * 8);
        *(uint4*)&G0[kc].u[0] = g0p[0]; *(uint4*)&G0[kc].u[4] = g0p[1];
        const uint4* g1p = (const uint4*)(gramB + (((size_t)kc * NTILES + nt1) * 32 + lane) * 8);
        *(uint4*)&G1[kc].u[0] = g1p[0]; *(uint4*)&G1[kc].u[4] = g1p[1];
    }

    int m_base = (lane >> 4) * 8;   // C-layout: VGPR v -> row v (lanes 0-15) / 8+v (lanes 16-31)
    int jin    = lane & 15;

    v8f u0 = {}, u1 = {};
    for (int it = 0; it < LCA_ITERS; it++) {
        // a = softshrink(u) -> bf16 -> LDS in A-fragment layout (K dim = 256 code dims)
        #pragma unroll
        for (int v = 0; v < 8; v++) {
            int m = m_base + v;
            {
                float a = softshrink(u0[v]);
                int j = nt0 * 16 + jin; int kc = j >> 5, kin = j & 31;
                aPack16[kc * 512 + a_local(m, kin)] = f2bf(a);
            }
            {
                float a = softshrink(u1[v]);
                int j = nt1 * 16 + jin; int kc = j >> 5, kin = j & 31;
                aPack16[kc * 512 + a_local(m, kin)] = f2bf(a);
            }
        }
        __syncthreads();

        v8f acc0 = {}, acc1 = {};
        #pragma unroll
        for (int kc = 0; kc < KCH_G; kc++) {
            Frag A;
            const uint4* ap = (const uint4*)(aPack + (kc * 32 + lane) * 8);
            *(uint4*)&A.u[0] = ap[0]; *(uint4*)&A.u[4] = ap[1];
            acc0 = wmma_bf16(A, G0[kc], acc0);
            acc1 = wmma_bf16(A, G1[kc], acc1);
        }
        // u += DT * (b - u - a@gram) / TAU
        u0 = u0 + DT_OVER_TAU * (b0 - u0 - acc0);
        u1 = u1 + DT_OVER_TAU * (b1 - u1 - acc1);
        __syncthreads();   // protect aPack WAR for next iteration
    }

    // codes = softshrink(u); max-pool over patches of each image
    #pragma unroll
    for (int v = 0; v < 8; v++) {
        int m   = m_base + v;
        int row = tile * 16 + m;
        int img = row / PPI;
        atomicMax(&out[img * N_BASES + nt0 * 16 + jin], softshrink(u0[v]));
        atomicMax(&out[img * N_BASES + nt1 * 16 + jin], softshrink(u1[v]));
    }
}

extern "C" void kernel_launch(void* const* d_in, const int* in_sizes, int n_in,
                              void* d_out, int out_size, void* d_ws, size_t ws_size,
                              hipStream_t stream) {
    (void)in_sizes; (void)n_in; (void)out_size; (void)ws_size;
    const float* image = (const float*)d_in[0];
    const float* dict  = (const float*)d_in[1];
    // d_in[2] = stride (8) hardcoded in the index math
    float* out = (float*)d_out;

    unsigned char* ws = (unsigned char*)d_ws;
    unsigned short* gramB16 = (unsigned short*)(ws + 0);          // 128 KB
    unsigned short* dictB16 = (unsigned short*)(ws + 131072);     // 384 KB
    unsigned short* pfA16   = (unsigned short*)(ws + 524288);     // 71.66 MB

    hipLaunchKernelGGL(gram_pack_kernel, dim3(256),       dim3(256), 0, stream, dict, gramB16);
    hipLaunchKernelGGL(dict_pack_kernel, dim3(INPUT_DIM), dim3(256), 0, stream, dict, dictB16);
    hipLaunchKernelGGL(patch_std_kernel, dim3(N_ROWS),    dim3(256), 0, stream, image, pfA16);
    hipLaunchKernelGGL(init_out_kernel,  dim3(64),        dim3(256), 0, stream, out);
    hipLaunchKernelGGL(lca_kernel,       dim3(ROW_TILES), dim3(256), 0, stream,
                       (const unsigned int*)pfA16, (const unsigned int*)dictB16,
                       (const unsigned int*)gramB16, out);
}
